// OneLayerAttention_64596308131859
// MI455X (gfx1250) — matrix operand order repaired and checked by reading, not compiled, not run on hardware
//
#include <hip/hip_runtime.h>
#include <hip/hip_bf16.h>

typedef __attribute__((ext_vector_type(16))) _Float16 v16h;
typedef __attribute__((ext_vector_type(8)))  _Float16 v8h;
typedef __attribute__((ext_vector_type(8)))  float    v8f;
typedef __attribute__((ext_vector_type(4)))  unsigned int v4u;
typedef __attribute__((ext_vector_type(8)))  unsigned int v8u;

// xor-lane exchange within wave32 via ds_swizzle (group-of-32 mode: xor in [14:10], and=0x1F)
#define SWZ_XOR(x, m) __int_as_float(__builtin_amdgcn_ds_swizzle(__float_as_int(x), (((m) & 31) << 10) | 0x1F))

__device__ __forceinline__ v8f wmma_f16(v16h a, v16h b, v8f c) {
  // D = A(16x32 f16) * B(32x16 f16) + C(16x16 f32)
  return __builtin_amdgcn_wmma_f32_16x16x32_f16(false, a, false, b, (short)0, c, false, false);
}

__device__ __forceinline__ v16h cat8(v8h lo, v8h hi8) {
  return __builtin_shufflevector(lo, hi8, 0, 1, 2, 3, 4, 5, 6, 7, 8, 9, 10, 11, 12, 13, 14, 15);
}

constexpr int Bsz    = 4;
constexpr int Sseq   = 4096;
constexpr int Dmodel = 1024;
constexpr int Hdim   = 128;
constexpr int BSrows = Bsz * Sseq;                 // 16384
constexpr float QSCALE = 0.08838834764831845f;     // 1/sqrt(128)

// ---------------------------------------------------------------------------
// TDM: issue a 2D tensor_load_to_lds of (nrows x 128) f16 elements, row stride
// 128, from global `gsrc` to LDS byte-offset `ldsOff`. Descriptor per CDNA5 ISA
// ch.8 (group0 128b, group1 256b). Wave-scalar op (EXEC ignored); call from one
// wave only. Tracked by TENSORcnt.
// ---------------------------------------------------------------------------
__device__ __forceinline__ void tdm_load_tile_f16(const _Float16* gsrc,
                                                  unsigned int ldsOff,
                                                  unsigned int nrows) {
  unsigned long long ga = (unsigned long long)(uintptr_t)gsrc;
  v4u g0;
  g0[0] = 1u;                                            // count=1 (valid), user mode
  g0[1] = ldsOff;                                        // lds_addr (bytes)
  g0[2] = (unsigned int)ga;                              // global_addr[31:0]
  g0[3] = (unsigned int)((ga >> 32) & 0x1FFFFFFu)        // global_addr[56:32]
        | (2u << 30);                                    // type = 2 ("image")
  v8u g1;
  g1[0] = (1u << 16);          // workgroup_mask=0 | data_size=1 (2 bytes)
  g1[1] = (128u << 16);        // tensor_dim0[15:0]=128 in [31:16]
  g1[2] = (nrows << 16);       // tensor_dim0[31:16]=0 | tensor_dim1[15:0]=nrows
  g1[3] = (128u << 16);        // tensor_dim1[31:16]=0 | tile_dim0=128
  g1[4] = nrows;               // tile_dim1=nrows | tile_dim2=0
  g1[5] = 128u;                // tensor_dim0_stride[31:0] = 128 elements
  g1[6] = 0u;                  // tensor_dim0_stride[47:32]=0 | dim1_stride lo=0
  g1[7] = 0u;                  // tensor_dim1_stride hi = 0
  asm volatile("tensor_load_to_lds %0, %1" :: "s"(g0), "s"(g1) : "memory");
}

// ---------------------------------------------------------------------------
// f32 -> f16 conversion (grid-stride)
// ---------------------------------------------------------------------------
__global__ void cvt_f32_to_f16(const float* __restrict__ src, _Float16* __restrict__ dst, long n) {
  long i = (long)blockIdx.x * blockDim.x + threadIdx.x;
  long stride = (long)gridDim.x * blockDim.x;
  for (; i < n; i += stride) dst[i] = (_Float16)src[i];
}

// ---------------------------------------------------------------------------
// QKV projection: Y[z] = Xh[16384 x 1024] @ W[z][1024 x 128], f16 in, f16 out.
// One wave per 16-row x 128-col output strip: X loaded once per strip, the A
// fragment feeds 8 back-to-back WMMAs; W (256KB) streams from L2 every strip.
// ---------------------------------------------------------------------------
__global__ __launch_bounds__(32) void qkv_proj(const _Float16* __restrict__ Xh,
                                               const _Float16* __restrict__ W3,
                                               _Float16* __restrict__ Y3) {
  const int lane = threadIdx.x;
  const int rowA = lane & 15;          // A-fragment: row M = lane%16
  const int hi   = lane >> 4;          // half-wave select
  const int kb   = hi * 8;             // A-fragment K base offset
  const int ncol = lane & 15;          // C-fragment: column N = lane%16

  const _Float16* W = W3 + (size_t)blockIdx.z * Dmodel * Hdim;
  _Float16*       Y = Y3 + (size_t)blockIdx.z * BSrows * Hdim;
  const int tm = blockIdx.x;

  const _Float16* xrow = Xh + (size_t)(tm * 16 + rowA) * Dmodel;

  v8f acc[8];
#pragma unroll
  for (int t = 0; t < 8; t++)
#pragma unroll
    for (int i = 0; i < 8; i++) acc[t][i] = 0.0f;

  for (int kc = 0; kc < Dmodel; kc += 32) {
    // A: lane holds X[row][kc+kb .. +7] and X[row][kc+kb+16 .. +23]
    v8h a0 = *(const v8h*)(xrow + kc + kb);
    v8h a1 = *(const v8h*)(xrow + kc + kb + 16);
    v16h a = cat8(a0, a1);
    const _Float16* wr = W + (size_t)kc * Hdim;  // B: column N = lane%16, K in table order
#pragma unroll
    for (int t = 0; t < 8; t++) {
      v16h b;
#pragma unroll
      for (int s = 0; s < 8; s++) {
        b[s]     = wr[(size_t)(kb + s) * Hdim + t * 16 + ncol];
        b[s + 8] = wr[(size_t)(kb + 16 + s) * Hdim + t * 16 + ncol];
      }
      acc[t] = wmma_f16(a, b, acc[t]);
    }
  }
#pragma unroll
  for (int t = 0; t < 8; t++)
#pragma unroll
    for (int r = 0; r < 8; r++)
      Y[(size_t)(tm * 16 + hi * 8 + r) * Hdim + t * 16 + ncol] = (_Float16)acc[t][r];
}

// ---------------------------------------------------------------------------
// Causal flash attention. Workgroup = 4 waves = 64 queries (one 16-query tile
// per wave). K/V tiles (32 keys x 128 dims f16) are staged into LDS by the
// Tensor Data Mover, double-buffered so the DMA overlaps WMMA compute; wave 0
// issues the TDM descriptors and gates the group with s_wait_tensorcnt +
// workgroup barrier. Computes S^T = K*Q^T so softmax stats are per-lane, and
// the exp'd score C-fragments concatenate directly into the P A-fragment.
// ---------------------------------------------------------------------------
__global__ __launch_bounds__(128) void flash_attn(const _Float16* __restrict__ Qh,
                                                  const _Float16* __restrict__ Kh,
                                                  const _Float16* __restrict__ Vh,
                                                  float* __restrict__ Out) {
  const int tid  = threadIdx.x;
  const int lane = tid & 31;
  const int w    = tid >> 5;           // wave id within workgroup (0..3)
  const int rowA = lane & 15;
  const int hi   = lane >> 4;
  const int kb   = hi * 8;
  const int ncol = lane & 15;

  const int b   = blockIdx.y;
  const int qtg = blockIdx.x;          // group of 4 query tiles
  const int qt  = qtg * 4 + w;         // this wave's query tile
  const size_t boff = (size_t)b * Sseq * Hdim;
  const _Float16* Q = Qh + boff;
  const _Float16* K = Kh + boff;
  const _Float16* V = Vh + boff;
  float*          O = Out + boff;

  __shared__ __align__(64) _Float16 sK[2][32 * Hdim];   // 2 x 8KB, TDM-filled
  __shared__ __align__(64) _Float16 sV[2][32 * Hdim];   // 2 x 8KB, TDM-filled
  __shared__ __align__(64) _Float16 sQt[4][Hdim * 16];  // per-wave Q^T, 4 x 4KB
  __shared__ float sBr[4][16];                          // per-wave broadcast

  // ---- stage this wave's Q^T (scaled by 1/sqrt(dk)) once ----
  {
    const int q  = lane & 15;
    const int d0 = hi * 64;            // two half-waves split the 128 dims
    const _Float16* qr = Q + (size_t)(qt * 16 + q) * Hdim + d0;
#pragma unroll
    for (int d = 0; d < 64; d += 8) {
      v8h t = *(const v8h*)(qr + d);
#pragma unroll
      for (int u = 0; u < 8; u++)
        sQt[w][q * Hdim + d0 + d + u] = (_Float16)((float)t[u] * QSCALE);
    }
  }

  const int kend  = (qt + 1) * 16;     // this wave's causal key bound
  const int kendg = (qtg + 1) * 64;    // workgroup key bound (uniform)

  // ---- prime the TDM pipeline: tile 0 into buffer 0 ----
  if (w == 0) {
    tdm_load_tile_f16(K, (unsigned int)(uintptr_t)&sK[0][0], 32u);
    tdm_load_tile_f16(V, (unsigned int)(uintptr_t)&sV[0][0], 32u);
  }
  __syncthreads();   // also covers sQt staging visibility

  // B-fragments of Q^T, resident in registers for the whole key loop
  v16h bQ[4];
#pragma unroll
  for (int c = 0; c < 4; c++) bQ[c] = cat8(*(const v8h*)(&sQt[w][ncol * Hdim + c * 32 + kb]), *(const v8h*)(&sQt[w][ncol * Hdim + c * 32 + kb + 16]));

  v8f o[8];
#pragma unroll
  for (int t = 0; t < 8; t++)
#pragma unroll
    for (int i = 0; i < 8; i++) o[t][i] = 0.0f;

  float mrun = -__builtin_inff();
  float lrun = 0.0f;
  const int myq = qt * 16 + ncol;      // query index owned by this lane

  for (int j0 = 0; j0 < kendg; j0 += 32) {
    const int cur = (j0 >> 5) & 1;
    // ---- prefetch next tile into the other buffer, then gate current ----
    if (w == 0) {
      if (j0 + 32 < kendg) {
        const int nxt = cur ^ 1;
        tdm_load_tile_f16(K + (size_t)(j0 + 32) * Hdim, (unsigned int)(uintptr_t)&sK[nxt][0], 32u);
        tdm_load_tile_f16(V + (size_t)(j0 + 32) * Hdim, (unsigned int)(uintptr_t)&sV[nxt][0], 32u);
        __builtin_amdgcn_s_wait_tensorcnt(2);  // current pair complete (in-order)
      } else {
        __builtin_amdgcn_s_wait_tensorcnt(0);
      }
    }
    __syncthreads();

    if (j0 < kend) {  // wave-uniform causal skip (EXEC stays all-ones)
      // ---- scores^T: two 16-key x 16-query tiles, K-dim 128 in 4 chunks ----
      v8f sc0, sc1;
#pragma unroll
      for (int i = 0; i < 8; i++) { sc0[i] = 0.0f; sc1[i] = 0.0f; }

#pragma unroll
      for (int c = 0; c < 4; c++) {
        const _Float16* kr = &sK[cur][(size_t)rowA * Hdim + c * 32 + kb];
        v16h aK = cat8(*(const v8h*)kr, *(const v8h*)(kr + 16));
        sc0 = wmma_f16(aK, bQ[c], sc0);
      }
#pragma unroll
      for (int c = 0; c < 4; c++) {
        const _Float16* kr = &sK[cur][(size_t)(16 + rowA) * Hdim + c * 32 + kb];
        v16h aK = cat8(*(const v8h*)kr, *(const v8h*)(kr + 16));
        sc1 = wmma_f16(aK, bQ[c], sc1);
      }

      // ---- causal mask + per-query (per-lane) online softmax ----
      float s0[8], s1[8];
      float rmax = -__builtin_inff();
#pragma unroll
      for (int r = 0; r < 8; r++) {
        const int key0 = j0 + hi * 8 + r;       // key index of C row r (tile 0)
        const int key1 = key0 + 16;             // tile 1
        float x0 = (key0 <= myq) ? sc0[r] : -__builtin_inff();
        float x1 = (key1 <= myq) ? sc1[r] : -__builtin_inff();
        s0[r] = x0; s1[r] = x1;
        rmax = fmaxf(rmax, fmaxf(x0, x1));
      }
      rmax = fmaxf(rmax, SWZ_XOR(rmax, 16));    // combine lane / lane^16 halves

      const float mnew  = fmaxf(mrun, rmax);
      const float alpha = __expf(mrun - mnew);

      float p0[8], p1[8], psum = 0.0f;
#pragma unroll
      for (int r = 0; r < 8; r++) {
        p0[r] = __expf(s0[r] - mnew);           // masked entries -> exp(-inf) = 0
        p1[r] = __expf(s1[r] - mnew);
        psum += p0[r] + p1[r];
      }
      psum += SWZ_XOR(psum, 16);
      lrun = lrun * alpha + psum;
      mrun = mnew;

      // broadcast alpha(query) to the lanes holding that query as a C-row
      if (lane < 16) sBr[w][lane] = alpha;      // single-wave LDS: DS ops in-order
      float arow[8];
#pragma unroll
      for (int r = 0; r < 8; r++) arow[r] = sBr[w][hi * 8 + r];

      // P A-fragment: C-fragment halves concatenate exactly into A layout
      v16h aP;
#pragma unroll
      for (int j = 0; j < 8; j++) {
        aP[j]     = (_Float16)p0[j];
        aP[j + 8] = (_Float16)p1[j];
      }

      // ---- O = diag(alpha)*O + P * V ; V B-fragments from LDS (lane = dim column, elements = keys) ----
#pragma unroll
      for (int t = 0; t < 8; t++) {
        v16h bV;
#pragma unroll
        for (int j = 0; j < 8; j++) {
          bV[j]     = sV[cur][(size_t)(kb + j) * Hdim + t * 16 + ncol];
          bV[j + 8] = sV[cur][(size_t)(kb + 16 + j) * Hdim + t * 16 + ncol];
        }
#pragma unroll
        for (int i = 0; i < 8; i++) o[t][i] *= arow[i];
        o[t] = wmma_f16(aP, bV, o[t]);
      }
    }
    __syncthreads();   // all reads of buffer `cur` done before it is refilled
  }

  // ---- epilogue: normalize by l and store f32 ----
  if (lane < 16) sBr[w][lane] = lrun;
  __syncthreads();
#pragma unroll
  for (int r = 0; r < 8; r++) {
    const float inv = 1.0f / sBr[w][hi * 8 + r];
#pragma unroll
    for (int t = 0; t < 8; t++)
      O[(size_t)(qt * 16 + hi * 8 + r) * Hdim + t * 16 + ncol] = o[t][r] * inv;
  }
}

// ---------------------------------------------------------------------------
// Host launcher
// ---------------------------------------------------------------------------
extern "C" void kernel_launch(void* const* d_in, const int* in_sizes, int n_in,
                              void* d_out, int out_size, void* d_ws, size_t ws_size,
                              hipStream_t stream) {
  const float* X  = (const float*)d_in[0];
  const float* Wq = (const float*)d_in[1];
  const float* Wk = (const float*)d_in[2];
  const float* Wv = (const float*)d_in[3];

  // workspace layout (halves): Xh | W3 (q,k,v) | Qh | Kh | Vh   (~45 MB total)
  _Float16* Xh = (_Float16*)d_ws;
  _Float16* W3 = Xh + (size_t)BSrows * Dmodel;
  _Float16* Q3 = W3 + 3 * (size_t)Dmodel * Hdim;
  _Float16* Qh = Q3;
  _Float16* Kp = Q3 + (size_t)BSrows * Hdim;
  _Float16* Vp = Kp + (size_t)BSrows * Hdim;

  cvt_f32_to_f16<<<4096, 256, 0, stream>>>(X, Xh, (long)BSrows * Dmodel);
  cvt_f32_to_f16<<<128, 256, 0, stream>>>(Wq, W3,                             (long)Dmodel * Hdim);
  cvt_f32_to_f16<<<128, 256, 0, stream>>>(Wk, W3 + (size_t)Dmodel * Hdim,     (long)Dmodel * Hdim);
  cvt_f32_to_f16<<<128, 256, 0, stream>>>(Wv, W3 + 2 * (size_t)Dmodel * Hdim, (long)Dmodel * Hdim);

  dim3 gp(BSrows / 16, 1, 3);          // 1024 row-strips x 3 matrices
  qkv_proj<<<gp, 32, 0, stream>>>(Xh, W3, Q3);

  dim3 ga(Sseq / 64, Bsz);             // 64 query-groups x 4 batches
  flash_attn<<<ga, 128, 0, stream>>>(Qh, Kp, Vp, (float*)d_out);
}
